// OIMLossCQ_9105330667998
// MI455X (gfx1250) — matrix operand order repaired and checked by reading, not compiled, not run on hardware
//
#include <hip/hip_runtime.h>
#include <hip/hip_bf16.h>
#include <math.h>

// Problem constants (match reference)
#define B_   4096
#define D_   512
#define Q_   16384
#define U_   256
#define OIM_ 30.0f
#define NCHUNK 8
#define QCHUNK (Q_ / NCHUNK)     // 2048 columns per workgroup
#define MTILE 32                 // rows per workgroup

typedef __attribute__((ext_vector_type(16))) __bf16 v16bf;
typedef __attribute__((ext_vector_type(8)))  float  v8f;

#define NEGBIG  (-1.0e30f)       // running-max init
#define MASKVAL (-2.0e30f)       // masked-column sentinel: exp(MASKVAL - NEGBIG) == 0

// ---------------- small helper kernels ----------------

__global__ void k_init(float* sums, int* cnt, float* out) {
    int i = blockIdx.x * 256 + threadIdx.x;
    if (i < U_ * D_) sums[i] = 0.0f;
    if (i < B_) cnt[i] = 0;
    if (i == 0) out[0] = 0.0f;
}

__global__ void k_hist(const int* __restrict__ labels, int* __restrict__ cnt) {
    int i = blockIdx.x * 256 + threadIdx.x;
    if (i < B_) {
        int v = labels[i];
        if ((unsigned)v < (unsigned)B_) atomicAdd(&cnt[v], 1);
    }
}

// sorted unique (by value) + value->unique-index map; tiny, single thread
__global__ void k_unique(const int* __restrict__ cnt, int* __restrict__ uniq,
                         int* __restrict__ vmap) {
    if (threadIdx.x == 0 && blockIdx.x == 0) {
        int u = 0;
        for (int v = 0; v < B_; ++v) {
            if (cnt[v] > 0) { uniq[u] = v; vmap[v] = u; ++u; }
        }
    }
}

// L2-normalize each input row, store bf16
__global__ void __launch_bounds__(128)
k_norm_inputs(const float* __restrict__ x, __hip_bfloat16* __restrict__ xn) {
    __shared__ float red[128];
    int row = blockIdx.x, tid = threadIdx.x;
    float s = 0.0f;
    for (int d = tid; d < D_; d += 128) { float v = x[(size_t)row * D_ + d]; s += v * v; }
    red[tid] = s; __syncthreads();
    for (int off = 64; off > 0; off >>= 1) {
        if (tid < off) red[tid] += red[tid + off];
        __syncthreads();
    }
    float inv = 1.0f / fmaxf(sqrtf(red[0]), 1e-12f);
    for (int d = tid; d < D_; d += 128)
        xn[(size_t)row * D_ + d] = __float2bfloat16(x[(size_t)row * D_ + d] * inv);
}

// accumulate per-pid sums
__global__ void __launch_bounds__(128)
k_sums(const float* __restrict__ x, const int* __restrict__ labels,
       const int* __restrict__ vmap, float* __restrict__ sums) {
    int row = blockIdx.x;
    int u = vmap[labels[row]];
    for (int d = threadIdx.x; d < D_; d += 128)
        atomicAdd(&sums[(size_t)u * D_ + d], x[(size_t)row * D_ + d]);
}

// mean + normalize -> uniq_emb_n (fp32)
__global__ void __launch_bounds__(128)
k_uniq_norm(const float* __restrict__ sums, const int* __restrict__ cnt,
            const int* __restrict__ uniq, float* __restrict__ un) {
    __shared__ float red[128];
    int u = blockIdx.x, tid = threadIdx.x;
    float c = (float)cnt[uniq[u]];
    float s = 0.0f;
    for (int d = tid; d < D_; d += 128) {
        float v = sums[(size_t)u * D_ + d] / c;
        s += v * v;
    }
    red[tid] = s; __syncthreads();
    for (int off = 64; off > 0; off >>= 1) {
        if (tid < off) red[tid] += red[tid + off];
        __syncthreads();
    }
    float inv = 1.0f / fmaxf(sqrtf(red[0]), 1e-12f);
    for (int d = tid; d < D_; d += 128)
        un[(size_t)u * D_ + d] = (sums[(size_t)u * D_ + d] / c) * inv;
}

// copy queue to workspace: emb fp32->bf16, labels
__global__ void k_copy_emb(const float* __restrict__ emb, const int* __restrict__ lab,
                           __hip_bfloat16* __restrict__ embW, int* __restrict__ labW) {
    size_t base = ((size_t)blockIdx.x * 256 + threadIdx.x) * 4;
    if (base + 3 < (size_t)Q_ * D_) {
        const float4 v = *(const float4*)(emb + base);
        embW[base + 0] = __float2bfloat16(v.x);
        embW[base + 1] = __float2bfloat16(v.y);
        embW[base + 2] = __float2bfloat16(v.z);
        embW[base + 3] = __float2bfloat16(v.w);
    }
    int j = blockIdx.x * 256 + threadIdx.x;
    if (j < Q_) labW[j] = lab[j];
}

// sequential circular queue update (exact scan semantics); one workgroup
__global__ void __launch_bounds__(256)
k_queue(const int* __restrict__ uniq, const float* __restrict__ un,
        const int* __restrict__ hdr, __hip_bfloat16* __restrict__ embW,
        int* __restrict__ labW) {
    __shared__ int sMin;
    int tid = threadIdx.x;
    int h0 = hdr[0];
    for (int s = 0; s < U_; ++s) {
        int y = uniq[s];
        if (tid == 0) sMin = 0x7fffffff;
        __syncthreads();
        for (int q = tid; q < Q_; q += 256)
            if (labW[q] == y) atomicMin(&sMin, q);
        __syncthreads();
        int i = sMin;
        int hh = (h0 + s) % Q_;
        for (int d = tid; d < D_; d += 256)
            embW[(size_t)hh * D_ + d] = __float2bfloat16(un[(size_t)s * D_ + d]);
        if (tid == 0) {
            labW[hh] = y;
            if (i != 0x7fffffff && i != hh) labW[i] = -1;   // invalidate stale slot
        }
        __syncthreads();
    }
}

// per-row target logit: the good slot for batch pid uniq[u] is (h0+u)%Q,
// holding exactly bf16(uniq_emb_n[u]) — same operands the GEMM consumes.
__global__ void __launch_bounds__(64)
k_tgt(const __hip_bfloat16* __restrict__ inN, const __hip_bfloat16* __restrict__ embW,
      const int* __restrict__ labels, const int* __restrict__ vmap,
      const int* __restrict__ hdr, float* __restrict__ tgt) {
    __shared__ float red[64];
    int b = blockIdx.x, tid = threadIdx.x;
    int u = vmap[labels[b]];
    int slot = (hdr[0] + u) % Q_;
    float s = 0.0f;
    for (int d = tid; d < D_; d += 64)
        s += __bfloat162float(inN[(size_t)b * D_ + d]) *
             __bfloat162float(embW[(size_t)slot * D_ + d]);
    red[tid] = s; __syncthreads();
    for (int off = 32; off > 0; off >>= 1) {
        if (tid < off) red[tid] += red[tid + off];
        __syncthreads();
    }
    if (tid == 0) tgt[b] = OIM_ * red[0];
}

// ---------------- fused WMMA GEMM + streaming log-sum-exp ----------------

#define AROWB 80                 // padded row pitch (bytes): 16B-aligned, conflict-free
#define ABUF_B (32 * AROWB)      // 2560 B per A buffer (32 rows x 64B payload)
#define BBUF_B (256 * AROWB)     // 20480 B per B buffer (256 rows x 64B payload)

__device__ __forceinline__ unsigned ldsOff(const void* p) {
    // low 32 bits of a generic shared-space address == LDS byte offset (ISA 10.2)
    return (unsigned)(unsigned long long)p;
}

// async-copy one 64-byte row (per lane) global -> LDS; offset applies to both sides
__device__ __forceinline__ void asyncRow64(unsigned ldsdst, const void* g) {
    asm volatile(
        "global_load_async_to_lds_b128 %0, %1, off\n\t"
        "global_load_async_to_lds_b128 %0, %1, off offset:16\n\t"
        "global_load_async_to_lds_b128 %0, %1, off offset:32\n\t"
        "global_load_async_to_lds_b128 %0, %1, off offset:48"
        :: "v"(ldsdst), "v"((unsigned long long)g) : "memory");
}

__device__ __forceinline__ void waitAsync0() {
    asm volatile("s_wait_asynccnt 0" ::: "memory");
}

// grid: (B_/MTILE, NCHUNK), block 256 (8 wave32: wm=wave&1 row-half, wn=wave>>1 col-quarter)
__global__ void __launch_bounds__(256)
k_gemm(const unsigned short* __restrict__ An,   // inputs_n bf16 [B,D]
       const unsigned short* __restrict__ Eb,   // queue emb bf16 [Q,D]
       const int* __restrict__ labq,            // updated queue labels [Q]
       float* __restrict__ partM, float* __restrict__ partS) {
    __shared__ __align__(16) unsigned char smem[2 * ABUF_B + 2 * BBUF_B + 1024]; // 47104 B

    const int tid  = threadIdx.x;
    const int lane = tid & 31;
    const int wave = tid >> 5;
    const int wm = wave & 1;
    const int wn = wave >> 1;
    const int half = lane >> 4;
    const int ncol = lane & 15;
    const int r0 = blockIdx.x * MTILE;
    const int q0 = blockIdx.y * QCHUNK;
    const int mrow = wm * 16 + ncol;     // A-matrix row this lane feeds

    float lmax[8], lsum[8];
#pragma unroll
    for (int r = 0; r < 8; ++r) { lmax[r] = NEGBIG; lsum[r] = 0.0f; }

    const int TOT = (QCHUNK / 256) * (D_ / 32);   // 128 tiles (ns-major, ks-minor)

    // prologue: issue tile 0 into buffer 0
    if (tid < 32)
        asyncRow64(ldsOff(smem + tid * AROWB),
                   An + (size_t)(r0 + tid) * D_);
    asyncRow64(ldsOff(smem + 2 * ABUF_B + tid * AROWB),
               Eb + (size_t)(q0 + tid) * D_);

    const v8f vz = {0.f, 0.f, 0.f, 0.f, 0.f, 0.f, 0.f, 0.f};

    for (int ns = 0; ns < QCHUNK / 256; ++ns) {
        const int qn = q0 + ns * 256;
        v8f acc[4];
#pragma unroll
        for (int t = 0; t < 4; ++t) acc[t] = vz;

        for (int ki = 0; ki < 16; ++ki) {
            waitAsync0();        // my async loads for this tile have landed
            __syncthreads();     // everyone's have landed; prior compute finished

            const int jn = ns * 16 + ki + 1;     // next tile (software pipeline)
            if (jn < TOT) {
                const int ks2 = (jn & 15) << 5;
                const int qn2 = q0 + (jn >> 4) * 256;
                const int p2 = jn & 1;
                if (tid < 32)
                    asyncRow64(ldsOff(smem + p2 * ABUF_B + tid * AROWB),
                               An + (size_t)(r0 + tid) * D_ + ks2);
                asyncRow64(ldsOff(smem + 2 * ABUF_B + p2 * BBUF_B + tid * AROWB),
                           Eb + (size_t)(qn2 + tid) * D_ + ks2);
            }

            const int p = ki & 1;                // (ns*16+ki)&1 == ki&1
            const unsigned char* aBuf = smem + p * ABUF_B;
            const unsigned char* bBuf = smem + 2 * ABUF_B + p * BBUF_B;

            // A fragment in contiguous k order (k = 16*half + s), matching B
            union { v16bf v; unsigned u[8]; } fa;
            const unsigned char* Ab = aBuf + mrow * AROWB;
#pragma unroll
            for (int v = 0; v < 8; ++v) {
                const int kb = v * 4 + half * 32;
                fa.u[v] = *(const unsigned*)(Ab + kb);
            }
#pragma unroll
            for (int t = 0; t < 4; ++t) {
                union { v16bf v; unsigned u[8]; } fb;
                const unsigned char* Bb = bBuf + (wn * 64 + t * 16 + ncol) * AROWB + half * 32;
#pragma unroll
                for (int v = 0; v < 8; ++v)
                    fb.u[v] = *(const unsigned*)(Bb + v * 4);
                acc[t] = __builtin_amdgcn_wmma_f32_16x16x32_bf16(
                    false, fa.v, false, fb.v, (short)0, acc[t], false, false);
            }
        }

        // fold this 256-column step into the running log-sum-exp (branch-free)
#pragma unroll
        for (int t = 0; t < 4; ++t) {
            const int q = qn + wn * 64 + t * 16 + ncol;
            const bool good = (labq[q] != -1);
#pragma unroll
            for (int r = 0; r < 8; ++r) {
                float val = good ? OIM_ * acc[t][r] : MASKVAL;
                float nM = fmaxf(lmax[r], val);
                lsum[r] = lsum[r] * __expf(lmax[r] - nM) + __expf(val - nM);
                lmax[r] = nM;
            }
        }
    }

    // reduce across the 16 column-lanes of each half-wave
#pragma unroll
    for (int r = 0; r < 8; ++r) {
#pragma unroll
        for (int m = 8; m >= 1; m >>= 1) {
            float oM = __shfl_xor(lmax[r], m, 16);
            float oS = __shfl_xor(lsum[r], m, 16);
            float M = fmaxf(lmax[r], oM);
            float S = lsum[r] * __expf(lmax[r] - M) + oS * __expf(oM - M);
            lmax[r] = M; lsum[r] = S;
        }
    }
    __syncthreads();
    {
        float* sM = (float*)(smem + 2 * ABUF_B + 2 * BBUF_B);   // [4][32]
        float* sS = sM + 128;                                    // [4][32]
        if ((lane & 15) == 0) {
#pragma unroll
            for (int r = 0; r < 8; ++r) {
                int rl = wm * 16 + half * 8 + r;
                sM[wn * 32 + rl] = lmax[r];
                sS[wn * 32 + rl] = lsum[r];
            }
        }
        __syncthreads();
        if (tid < 32) {
            float M = NEGBIG, S = 0.0f;
#pragma unroll
            for (int w = 0; w < 4; ++w) {
                float m2 = sM[w * 32 + tid], s2 = sS[w * 32 + tid];
                float nM = fmaxf(M, m2);
                S = S * __expf(M - nM) + s2 * __expf(m2 - nM);
                M = nM;
            }
            partM[(size_t)(r0 + tid) * NCHUNK + blockIdx.y] = M;
            partS[(size_t)(r0 + tid) * NCHUNK + blockIdx.y] = S;
        }
    }
}

// combine chunk partials into the scalar loss: mean(M + log S - t)
__global__ void __launch_bounds__(256)
k_loss(const float* __restrict__ pM, const float* __restrict__ pS,
       const float* __restrict__ tgt, float* __restrict__ out) {
    __shared__ float red[256];
    int row = blockIdx.x * 256 + threadIdx.x;
    float M = NEGBIG, S = 0.0f;
#pragma unroll
    for (int c = 0; c < NCHUNK; ++c) {
        float m2 = pM[(size_t)row * NCHUNK + c];
        float s2 = pS[(size_t)row * NCHUNK + c];
        float nM = fmaxf(M, m2);
        S = S * __expf(M - nM) + s2 * __expf(m2 - nM);
        M = nM;
    }
    float rl = M + logf(S) - tgt[row];
    red[threadIdx.x] = rl;
    __syncthreads();
    for (int off = 128; off > 0; off >>= 1) {
        if (threadIdx.x < off) red[threadIdx.x] += red[threadIdx.x + off];
        __syncthreads();
    }
    if (threadIdx.x == 0) atomicAdd(out, red[0] * (1.0f / (float)B_));
}

// ---------------- host launcher ----------------

extern "C" void kernel_launch(void* const* d_in, const int* in_sizes, int n_in,
                              void* d_out, int out_size, void* d_ws, size_t ws_size,
                              hipStream_t stream) {
    (void)in_sizes; (void)n_in; (void)out_size; (void)ws_size;
    const float* inputs   = (const float*)d_in[0];
    const int*   labels   = (const int*)d_in[1];
    const float* emb_cq   = (const float*)d_in[2];
    const int*   label_cq = (const int*)d_in[3];
    // d_in[4] = age_cq: updated in reference but dead for the loss output
    const int*   header   = (const int*)d_in[5];
    float* out = (float*)d_out;

    char* ws = (char*)d_ws;
    size_t o = 0;
    auto take = [&](size_t bytes) -> char* {
        char* p = ws + o;
        o = (o + bytes + 255) & ~(size_t)255;
        return p;
    };
    __hip_bfloat16* inN  = (__hip_bfloat16*)take((size_t)B_ * D_ * 2);
    __hip_bfloat16* embW = (__hip_bfloat16*)take((size_t)Q_ * D_ * 2);
    int*   labW  = (int*)take((size_t)Q_ * 4);
    float* sums  = (float*)take((size_t)U_ * D_ * 4);
    int*   cnt   = (int*)take((size_t)B_ * 4);
    int*   uniq  = (int*)take((size_t)U_ * 4);
    int*   vmap  = (int*)take((size_t)B_ * 4);
    float* uniqN = (float*)take((size_t)U_ * D_ * 4);
    float* pM    = (float*)take((size_t)B_ * NCHUNK * 4);
    float* pS    = (float*)take((size_t)B_ * NCHUNK * 4);
    float* tgt   = (float*)take((size_t)B_ * 4);

    k_init<<<512, 256, 0, stream>>>(sums, cnt, out);
    k_hist<<<16, 256, 0, stream>>>(labels, cnt);
    k_unique<<<1, 32, 0, stream>>>(cnt, uniq, vmap);
    k_norm_inputs<<<B_, 128, 0, stream>>>(inputs, inN);
    k_sums<<<B_, 128, 0, stream>>>(inputs, labels, vmap, sums);
    k_uniq_norm<<<U_, 128, 0, stream>>>(sums, cnt, uniq, uniqN);
    k_copy_emb<<<8192, 256, 0, stream>>>(emb_cq, label_cq, embW, labW);
    k_queue<<<1, 256, 0, stream>>>(uniq, uniqN, header, embW, labW);
    k_tgt<<<B_, 64, 0, stream>>>(inN, embW, labels, vmap, header, tgt);
    dim3 g(B_ / MTILE, NCHUNK);
    k_gemm<<<g, 256, 0, stream>>>((const unsigned short*)inN,
                                  (const unsigned short*)embW,
                                  labW, pM, pS);
    k_loss<<<B_ / 256, 256, 0, stream>>>(pM, pS, tgt, out);
}
